// S_Attention_55327768708399
// MI455X (gfx1250) — matrix operand order repaired and checked by reading, not compiled, not run on hardware
//
#include <hip/hip_runtime.h>
#include <math.h>

typedef _Float16 f16;
typedef __attribute__((ext_vector_type(8)))  f16   v8h;
typedef __attribute__((ext_vector_type(16))) f16   v16h;
typedef __attribute__((ext_vector_type(8)))  float v8f;

union F16x16 { v16h v; v8h h[2]; f16 e[16]; };

#define B_  8
#define N_  1025
#define C_  512
#define H_  8
#define HD_ 64
#define NP_ 1025

__device__ __forceinline__ v8f wmma16(v16h a, v16h b, v8f c) {
  // D(f32 16x16) = A(f16 16x32) * B(f16 32x16) + C
  return __builtin_amdgcn_wmma_f32_16x16x32_f16(false, a, false, b, (short)0, c,
                                                false, false);
}

// ---------------------------------------------------------------------------
// Kernel 1: geometry bias  ->  bias[h][i][j], (H, 1025, 1025), pad row/col 0
// ---------------------------------------------------------------------------
__global__ void bias_kernel(const float* __restrict__ wg_w,
                            const float* __restrict__ wg_b,
                            float* __restrict__ bias) {
  __shared__ float swg[H_ * 64];
  __shared__ float swb[H_];
  int t = threadIdx.x;
  for (int u = t; u < H_ * 64; u += 256) swg[u] = wg_w[u];
  if (t < H_) swb[t] = wg_b[t];
  __syncthreads();

  long idx = (long)blockIdx.x * 256 + t;
  if (idx >= (long)NP_ * NP_) return;
  int i = (int)(idx / NP_), j = (int)(idx % NP_);
  if (i == 0 || j == 0) {
    for (int h = 0; h < H_; ++h) bias[(long)h * NP_ * NP_ + idx] = 0.f;
    return;
  }
  int p = i - 1, q = j - 1;
  float cxi = ((p >> 5) + 0.5f) * (1.f / 32.f);
  float cyi = ((p & 31) + 0.5f) * (1.f / 32.f);
  float cxj = ((q >> 5) + 0.5f) * (1.f / 32.f);
  float cyj = ((q & 31) + 0.5f) * (1.f / 32.f);
  const float wv = 1.f / 32.f + 1.f;  // w == h, identical for every cell
  float pos[4];
  pos[0] = logf(fmaxf(fabsf((cxi - cxj) / wv), 0.001f));
  pos[1] = logf(fmaxf(fabsf((cyi - cyj) / wv), 0.001f));
  pos[2] = 0.f;  // log(w_i / w_j) == 0
  pos[3] = 0.f;

  float acc[H_];
  for (int h = 0; h < H_; ++h) acc[h] = swb[h];
  for (int td = 0; td < 4; ++td) {
    float base = 100.f * pos[td];
    for (int d = 0; d < 8; ++d) {
      float m = base * __powf(1000.f, -(float)d * 0.125f);
      float sm = sinf(m), cm = cosf(m);
      int k = td * 8 + d;
      for (int h = 0; h < H_; ++h)
        acc[h] += sm * swg[h * 64 + k] + cm * swg[h * 64 + 32 + k];
    }
  }
  for (int h = 0; h < H_; ++h) {
    float r = fmaxf(acc[h], 0.f);
    bias[(long)h * NP_ * NP_ + idx] = logf(fmaxf(r, 1e-6f));
  }
}

// ---------------------------------------------------------------------------
// Elementwise f32 -> f16 pre-conversion (one pass, bandwidth-trivial)
// ---------------------------------------------------------------------------
__global__ void cvt_f16_kernel(const float* __restrict__ src,
                               f16* __restrict__ dst, long n) {
  long i = (long)blockIdx.x * 256 + threadIdx.x;
  if (i < n) dst[i] = (f16)src[i];
}

// ---------------------------------------------------------------------------
// GEMM: out[m,o] = sum_k A[m,k]*W[o,k]  (A, W in f16; accumulate f32)
// One 16x16 tile per wave, K stepped by 32 through v_wmma_f32_16x16x32_f16.
// outh != null -> store f16 (no bias); else store f32 + bias_v[o].
// ---------------------------------------------------------------------------
__global__ void gemm_nt_f16(const f16* __restrict__ A, const f16* __restrict__ W,
                            const float* __restrict__ bias_v,
                            float* __restrict__ outf, f16* __restrict__ outh,
                            int M, int K, int O) {
  int wave = threadIdx.x >> 5;
  int lane = threadIdx.x & 31;
  int half = lane >> 4;
  int l16  = lane & 15;
  int wid  = blockIdx.x * (blockDim.x >> 5) + wave;
  int OT = O >> 4;
  int MT = (M + 15) >> 4;
  if (wid >= MT * OT) return;
  int mt = wid / OT, ot = wid % OT;

  int ra = mt * 16 + l16; if (ra > M - 1) ra = M - 1;  // clamp edge rows
  int rw = ot * 16 + l16;
  const f16* Arow = A + (long)ra * K;
  const f16* Wrow = W + (long)rw * K;

  v8f acc = {};
  for (int k = 0; k < K; k += 32) {
    // A 16x32 fragment: lane row=l16; K = half*8+[0..7] and 16+half*8+[0..7]
    F16x16 af, bf;
    af.h[0] = *(const v8h*)(Arow + k + half * 8);
    af.h[1] = *(const v8h*)(Arow + k + 16 + half * 8);
    // B 32x16 fragment: lane col=l16; K = half*8+[0..7] and 16+half*8+[0..7]
    bf.h[0] = *(const v8h*)(Wrow + k + half * 8);
    bf.h[1] = *(const v8h*)(Wrow + k + 16 + half * 8);
    acc = wmma16(af.v, bf.v, acc);
  }

  int col = ot * 16 + l16;
  for (int r = 0; r < 8; ++r) {
    int row = mt * 16 + half * 8 + r;  // C layout: reg r <-> row r / r+8
    if (row < M) {
      if (outh) outh[(long)row * O + col] = (f16)acc[r];
      else      outf[(long)row * O + col] = acc[r] + bias_v[col];
    }
  }
}

// ---------------------------------------------------------------------------
// Flash attention: one (b, h, 16-row i-tile) per wave, j in tiles of 32 so
// P (16x32) is exactly one f16 A-fragment. f32 softmax + accumulation.
// ---------------------------------------------------------------------------
__global__ void attn_kernel(const f16* __restrict__ qkv,
                            const float* __restrict__ bias,
                            f16* __restrict__ ao) {
  __shared__ __align__(16) f16 pbuf[8][16 * 32];
  int wave = threadIdx.x >> 5;
  int lane = threadIdx.x & 31;
  int half = lane >> 4;
  int l16  = lane & 15;
  const int NTI = (N_ + 15) / 16;  // 65
  const int NTJ = (N_ + 31) / 32;  // 33
  int wid = blockIdx.x * 8 + wave;
  if (wid >= B_ * H_ * NTI) return;
  int bb  = wid / (H_ * NTI);
  int rem = wid % (H_ * NTI);
  int h   = rem / NTI;
  int it  = rem % NTI;
  int i0  = it * 16;

  // Q tile (16x64) as two A-fragments (K = 0..31, 32..63)
  v16h qa[2];
  {
    int ri = i0 + l16; if (ri > N_ - 1) ri = N_ - 1;
    const f16* qrow = qkv + (long)(bb * N_ + ri) * (3 * C_) + h * HD_;
    for (int s = 0; s < 2; ++s) {
      F16x16 f;
      f.h[0] = *(const v8h*)(qrow + s * 32 + half * 8);
      f.h[1] = *(const v8h*)(qrow + s * 32 + 16 + half * 8);
      qa[s] = f.v;
    }
  }
  const f16* kbase = qkv + (long)bb * N_ * 3 * C_ + C_ + h * HD_;
  const f16* vbase = qkv + (long)bb * N_ * 3 * C_ + 2 * C_ + h * HD_;
  const float* brow = bias + (long)h * NP_ * NP_;

  v8f oacc[4] = {};
  float mrun[8], lrun[8];
  for (int r = 0; r < 8; ++r) { mrun[r] = -__builtin_inff(); lrun[r] = 0.f; }
  const float scale = 0.125f;  // 64^-0.5

  for (int jt = 0; jt < NTJ; ++jt) {
    int j0 = jt * 32;
    // S (16x32) = Q K^T as two 16x16 C tiles, each 2 WMMAs over K=64
    v8f st0 = {}, st1 = {};
    for (int t = 0; t < 2; ++t) {
      int rk = j0 + t * 16 + l16; if (rk > N_ - 1) rk = N_ - 1;
      const f16* krow = kbase + (long)rk * (3 * C_);
      for (int s = 0; s < 2; ++s) {
        F16x16 f;  // B fragment: col=l16, K = s*32 + half*8 + [0..7] and s*32 + 16 + half*8 + [0..7]
        f.h[0] = *(const v8h*)(krow + s * 32 + half * 8);
        f.h[1] = *(const v8h*)(krow + s * 32 + 16 + half * 8);
        if (t == 0) st0 = wmma16(qa[s], f.v, st0);
        else        st1 = wmma16(qa[s], f.v, st1);
      }
    }
    int col0 = j0 + l16, col1 = j0 + 16 + l16;
    int c0c = col0 > N_ - 1 ? N_ - 1 : col0;
    int c1c = col1 > N_ - 1 ? N_ - 1 : col1;

    for (int r = 0; r < 8; ++r) {
      int row  = i0 + half * 8 + r;
      int rowc = row > N_ - 1 ? N_ - 1 : row;
      const float* bptr = brow + (long)rowc * NP_;
      float sv0 = st0[r] * scale + bptr[c0c];
      float sv1 = st1[r] * scale + bptr[c1c];
      if (col0 >= N_) sv0 = -__builtin_inff();
      if (col1 >= N_) sv1 = -__builtin_inff();
      float mx = fmaxf(sv0, sv1);
      mx = fmaxf(mx, __shfl_xor(mx, 1, 32));
      mx = fmaxf(mx, __shfl_xor(mx, 2, 32));
      mx = fmaxf(mx, __shfl_xor(mx, 4, 32));
      mx = fmaxf(mx, __shfl_xor(mx, 8, 32));
      float mnew  = fmaxf(mrun[r], mx);
      float alpha = __expf(mrun[r] - mnew);
      float p0 = __expf(sv0 - mnew);
      float p1 = __expf(sv1 - mnew);
      float ps = p0 + p1;
      ps += __shfl_xor(ps, 1, 32);
      ps += __shfl_xor(ps, 2, 32);
      ps += __shfl_xor(ps, 4, 32);
      ps += __shfl_xor(ps, 8, 32);
      lrun[r] = lrun[r] * alpha + ps;
      mrun[r] = mnew;
      for (int dt = 0; dt < 4; ++dt) oacc[dt][r] *= alpha;
      pbuf[wave][(half * 8 + r) * 32 + l16]      = (f16)p0;  // row-major 16x32
      pbuf[wave][(half * 8 + r) * 32 + 16 + l16] = (f16)p1;
    }
    // LDS in-order within a wave; wait stores, re-read P in A-fragment layout
    asm volatile("s_wait_dscnt 0" ::: "memory");
    F16x16 pf;
    pf.h[0] = *(const v8h*)(&pbuf[wave][l16 * 32 + half * 8]);
    pf.h[1] = *(const v8h*)(&pbuf[wave][l16 * 32 + 16 + half * 8]);
    // V fragments (32x16 per 16-col d-subtile): lane K = half*8 + [0..7] and 16 + half*8 + [0..7]
    F16x16 vf[4];
    for (int u = 0; u < 16; ++u) {
      int jr = j0 + (u < 8 ? half * 8 + u : 16 + half * 8 + (u - 8)); if (jr > N_ - 1) jr = N_ - 1;
      const f16* vrow = vbase + (long)jr * (3 * C_);
      for (int dt = 0; dt < 4; ++dt) vf[dt].e[u] = vrow[dt * 16 + l16];
    }
    for (int dt = 0; dt < 4; ++dt)
      oacc[dt] = wmma16(pf.v, vf[dt].v, oacc[dt]);
  }

  for (int r = 0; r < 8; ++r) {
    int row = i0 + half * 8 + r;
    if (row < N_) {
      float inv = 1.f / lrun[r];
      f16* orow = ao + (long)(bb * N_ + row) * C_ + h * HD_;
      for (int dt = 0; dt < 4; ++dt)
        orow[dt * 16 + l16] = (f16)(oacc[dt][r] * inv);
    }
  }
}

// ---------------------------------------------------------------------------
extern "C" void kernel_launch(void* const* d_in, const int* in_sizes, int n_in,
                              void* d_out, int out_size, void* d_ws,
                              size_t ws_size, hipStream_t stream) {
  const float* x      = (const float*)d_in[0];
  const float* qkv_w  = (const float*)d_in[1];
  const float* proj_w = (const float*)d_in[2];
  const float* proj_b = (const float*)d_in[3];
  const float* wg_w   = (const float*)d_in[4];
  const float* wg_b   = (const float*)d_in[5];
  float* out = (float*)d_out;

  // Workspace layout (all chunks 16B aligned)
  char* base = (char*)d_ws;
  float* bias = (float*)base;  base += (size_t)H_ * NP_ * NP_ * sizeof(float);
  f16* xh   = (f16*)base;      base += (size_t)(B_ * N_) * C_ * sizeof(f16);
  f16* wqh  = (f16*)base;      base += (size_t)(3 * C_) * C_ * sizeof(f16);
  f16* wph  = (f16*)base;      base += (size_t)C_ * C_ * sizeof(f16);
  f16* qkvh = (f16*)base;      base += (size_t)(B_ * N_) * (3 * C_) * sizeof(f16);
  f16* aoh  = (f16*)base;

  {  // geometry bias (fp32)
    long total = (long)NP_ * NP_;
    bias_kernel<<<(int)((total + 255) / 256), 256, 0, stream>>>(wg_w, wg_b,
                                                                bias);
  }
  {  // one-time f32 -> f16 conversions
    long nx = (long)B_ * N_ * C_;
    cvt_f16_kernel<<<(int)((nx + 255) / 256), 256, 0, stream>>>(x, xh, nx);
    long nq = (long)3 * C_ * C_;
    cvt_f16_kernel<<<(int)((nq + 255) / 256), 256, 0, stream>>>(qkv_w, wqh, nq);
    long np = (long)C_ * C_;
    cvt_f16_kernel<<<(int)((np + 255) / 256), 256, 0, stream>>>(proj_w, wph, np);
  }
  {  // QKV projection, f16 in/out, f32 accumulate
    int M = B_ * N_, K = C_, O = 3 * C_;
    int waves = ((M + 15) / 16) * (O / 16);
    gemm_nt_f16<<<(waves + 7) / 8, 256, 0, stream>>>(xh, wqh, nullptr, nullptr,
                                                     qkvh, M, K, O);
  }
  {  // flash attention
    int NTI = (N_ + 15) / 16;
    int waves = B_ * H_ * NTI;
    attn_kernel<<<(waves + 7) / 8, 256, 0, stream>>>(qkvh, bias, aoh);
  }
  {  // output projection + bias, f32 output
    int M = B_ * N_, K = C_, O = C_;
    int waves = ((M + 15) / 16) * (O / 16);
    gemm_nt_f16<<<(waves + 7) / 8, 256, 0, stream>>>(aoh, wph, proj_b, out,
                                                     nullptr, M, K, O);
  }
}
